// GlobalDCTFilter_1d_45775761441153
// MI455X (gfx1250) — matrix operand order repaired and checked by reading, not compiled, not run on hardware
//
#include <hip/hip_runtime.h>
#include <hip/hip_bf16.h>

typedef __attribute__((ext_vector_type(16))) __bf16 v16bf;
typedef __attribute__((ext_vector_type(8)))  float  v8f;

#define NREAL 196     // DCT length (14*14)
#define CDIM  768     // channels
#define BDIM  256     // batch
#define NP    224     // padded contraction length (7*32)
#define MR    208     // padded row count for D matrices / padded weight (13*16)
#define CHUNK 64      // channels per block
#define GB    4       // batches per block
#define THREADS 512
#define NWAVES  16
#define NTILE_M 13    // 13*16 = 208 covers 196 rows
#define NTILE_C 4     // 64/16
#define KITERS  7     // 224/32
#define XVECS   7     // ceil((CHUNK/4)*NREAL / THREADS) float4 prefetch regs per thread
#define PI_D 3.14159265358979323846

// ---------------- kernel 0a: generate D (DCT-II) and Dinv (DCT-III/N) in bf16 --------
__global__ void gen_dct_mats(__bf16* __restrict__ D, __bf16* __restrict__ Di) {
  int idx = blockIdx.x * blockDim.x + threadIdx.x;
  if (idx >= MR * NP) return;
  int r = idx / NP;   // row
  int c = idx % NP;   // col
  float d = 0.f, di = 0.f;
  if (r < NREAL && c < NREAL) {
    // D[k=r][n=c] = 2*cos(pi*k*(2n+1)/(2N))
    d = (float)(2.0 * cos((double)r * (double)(2 * c + 1) * (PI_D / (2.0 * NREAL))));
    // Dinv[n=r][k=c] = (ck/N)*cos(pi*k*(2n+1)/(2N)), c0 = 1/2
    double ck = (c == 0) ? 0.5 : 1.0;
    di = (float)((ck / (double)NREAL) *
                 cos((double)c * (double)(2 * r + 1) * (PI_D / (2.0 * NREAL))));
  }
  D[idx]  = (__bf16)d;
  Di[idx] = (__bf16)di;
}

// ---------------- kernel 0b: zero-padded weight Wp[c][k], k padded to MR=208 ---------
__global__ void gen_wpad(const float* __restrict__ wg, float* __restrict__ wp) {
  int idx = blockIdx.x * blockDim.x + threadIdx.x;
  if (idx >= CDIM * MR) return;
  int c = idx / MR;
  int k = idx % MR;
  wp[idx] = (k < NREAL) ? wg[c * NREAL + k] : 0.0f;
}

// ---------------- main fused kernel: y = Dinv * (W .* (D * x_slab)) ------------------
__global__ __launch_bounds__(THREADS) void dct_filter_main(
    const float* __restrict__ xg, const float* __restrict__ wp,
    const __bf16* __restrict__ Dg, const __bf16* __restrict__ Dig,
    float* __restrict__ yg)
{
  __shared__ __bf16 Dlds [MR * NP];     // D[k][n], n contiguous
  __shared__ __bf16 Dilds[MR * NP];     // Dinv[n][k], k contiguous
  __shared__ __bf16 xlds [CHUNK * NP];  // x slab transposed: [c_local][n]
  __shared__ __bf16 zlds [CHUNK * NP];  // Z' = W.*Z transposed: [c_local][k]

  const int tid  = threadIdx.x;
  const int wid  = tid >> 5;
  const int lane = tid & 31;
  const int l16  = lane & 15;
  const int h    = lane >> 4;
  const int c0    = blockIdx.x * CHUNK;
  const int bbase = blockIdx.y * GB;
  const int XTOT  = (CHUNK / 4) * NREAL;   // 3136 float4 transfers per slab

  // stage 0a: copy D/Dinv into LDS (float4 = 8 bf16 per transfer)
  {
    const float4* s0 = (const float4*)Dg;
    const float4* s1 = (const float4*)Dig;
    float4* d0 = (float4*)Dlds;
    float4* d1 = (float4*)Dilds;
    const int nv = MR * NP / 8;
    for (int i = tid; i < nv; i += THREADS) { d0[i] = s0[i]; d1[i] = s1[i]; }
  }
  // stage 0b: zero the pad regions once (loads only ever write n<196 / k<208)
  for (int i = tid; i < CHUNK * (NP - NREAL); i += THREADS) {
    int cc = i / (NP - NREAL);
    int nn = NREAL + i % (NP - NREAL);
    xlds[cc * NP + nn] = (__bf16)0.0f;
  }
  for (int i = tid; i < CHUNK * (NP - MR); i += THREADS) {
    int cc = i / (NP - MR);
    int kk = MR + i % (NP - MR);
    zlds[cc * NP + kk] = (__bf16)0.0f;
  }

  // prefetch batch 0's x slab into registers (global latency overlaps D/Dinv copy)
  float4 xr[XVECS];
  {
    const float4* x4 = (const float4*)(xg + (size_t)bbase * NREAL * CDIM + c0);
    #pragma unroll
    for (int i = 0; i < XVECS; ++i) {
      int idx = tid + i * THREADS;
      if (idx < XTOT) xr[i] = x4[(size_t)(idx >> 4) * (CDIM / 4) + (idx & 15)];
    }
  }
  __syncthreads();

  for (int g = 0; g < GB; ++g) {
    const int b = bbase + g;

    // commit prefetched slab: f32 regs -> bf16 transposed LDS [c_local][n]
    #pragma unroll
    for (int i = 0; i < XVECS; ++i) {
      int idx = tid + i * THREADS;
      if (idx < XTOT) {
        int n  = idx >> 4;
        int cl = (idx & 15) * 4;
        float4 v = xr[i];
        xlds[(cl + 0) * NP + n] = (__bf16)v.x;
        xlds[(cl + 1) * NP + n] = (__bf16)v.y;
        xlds[(cl + 2) * NP + n] = (__bf16)v.z;
        xlds[(cl + 3) * NP + n] = (__bf16)v.w;
      }
    }
    __syncthreads();

    // issue next batch's global loads now; they complete during stage 1/2 compute
    if (g + 1 < GB) {
      const float4* x4 = (const float4*)(xg + (size_t)(b + 1) * NREAL * CDIM + c0);
      #pragma unroll
      for (int i = 0; i < XVECS; ++i) {
        int idx = tid + i * THREADS;
        if (idx < XTOT) xr[i] = x4[(size_t)(idx >> 4) * (CDIM / 4) + (idx & 15)];
      }
    }

    // ---- stage 1: Z[k,c] = sum_n D[k,n] * x[n,c];  zlds[c][k] = bf16(Z * Wp[c,k]) ---
    for (int t = wid; t < NTILE_M * NTILE_C; t += NWAVES) {
      const int k0  = (t % NTILE_M) * 16;
      const int cc0 = (t / NTILE_M) * 16;
      v8f acc = {0.f, 0.f, 0.f, 0.f, 0.f, 0.f, 0.f, 0.f};
      const int arow = (k0  + l16) * NP;
      const int brow = (cc0 + l16) * NP;
      #pragma unroll
      for (int kk = 0; kk < KITERS; ++kk) {
        const int n0 = kk * 32;
        union { float4 f[2]; v16bf v; } A, Bm;
        // A fragment (16x32 bf16): lane K-chunks at 8h and 16+8h
        A.f[0]  = *(const float4*)(&Dlds[arow + n0 + 8 * h]);
        A.f[1]  = *(const float4*)(&Dlds[arow + n0 + 16 + 8 * h]);
        // B fragment (32x16 bf16): lane K-chunks at 8h and 16+8h
        Bm.f[0] = *(const float4*)(&xlds[brow + n0 + 8 * h]);
        Bm.f[1] = *(const float4*)(&xlds[brow + n0 + 16 + 8 * h]);
        acc = __builtin_amdgcn_wmma_f32_16x16x32_bf16(
            false, A.v, false, Bm.v, (short)0, acc, false, false);
      }
      // epilogue: each lane's 8 results are k = k0+8h+0..7 -> two float4 weight loads
      // (zero-padded for k>=196), pack 8 bf16, single ds_store_b128.
      const int c = c0 + cc0 + l16;
      const float4* wp4 = (const float4*)(wp + (size_t)c * MR + k0 + 8 * h);
      float4 w0 = wp4[0];
      float4 w1 = wp4[1];
      union { __bf16 e[8]; float4 f; } pk;
      pk.e[0] = (__bf16)(acc[0] * w0.x);
      pk.e[1] = (__bf16)(acc[1] * w0.y);
      pk.e[2] = (__bf16)(acc[2] * w0.z);
      pk.e[3] = (__bf16)(acc[3] * w0.w);
      pk.e[4] = (__bf16)(acc[4] * w1.x);
      pk.e[5] = (__bf16)(acc[5] * w1.y);
      pk.e[6] = (__bf16)(acc[6] * w1.z);
      pk.e[7] = (__bf16)(acc[7] * w1.w);
      *(float4*)(&zlds[(cc0 + l16) * NP + k0 + 8 * h]) = pk.f;
    }
    __syncthreads();

    // ---- stage 2: y[n,c] = sum_k Dinv[n,k] * Z'[k,c] -------------------------------
    for (int t = wid; t < NTILE_M * NTILE_C; t += NWAVES) {
      const int n0  = (t % NTILE_M) * 16;
      const int cc0 = (t / NTILE_M) * 16;
      v8f acc = {0.f, 0.f, 0.f, 0.f, 0.f, 0.f, 0.f, 0.f};
      const int arow = (n0  + l16) * NP;
      const int brow = (cc0 + l16) * NP;
      #pragma unroll
      for (int kk = 0; kk < KITERS; ++kk) {
        const int kc = kk * 32;
        union { float4 f[2]; v16bf v; } A, Bm;
        A.f[0]  = *(const float4*)(&Dilds[arow + kc + 8 * h]);
        A.f[1]  = *(const float4*)(&Dilds[arow + kc + 16 + 8 * h]);
        Bm.f[0] = *(const float4*)(&zlds[brow + kc + 8 * h]);
        Bm.f[1] = *(const float4*)(&zlds[brow + kc + 16 + 8 * h]);
        acc = __builtin_amdgcn_wmma_f32_16x16x32_bf16(
            false, A.v, false, Bm.v, (short)0, acc, false, false);
      }
      // write y[b, n, c] (f32, coalesced 64B runs). Fast path: full tile (12 of 13).
      const int c = c0 + cc0 + l16;
      float* yrow = yg + (size_t)b * NREAL * CDIM + c;
      if (n0 + 16 <= NREAL) {
        #pragma unroll
        for (int q = 0; q < 8; ++q)
          yrow[(size_t)(n0 + q + 8 * h) * CDIM] = acc[q];
      } else {
        #pragma unroll
        for (int q = 0; q < 8; ++q) {
          const int n = n0 + q + 8 * h;
          if (n < NREAL) yrow[(size_t)n * CDIM] = acc[q];
        }
      }
    }
    __syncthreads();
  }
}

extern "C" void kernel_launch(void* const* d_in, const int* in_sizes, int n_in,
                              void* d_out, int out_size, void* d_ws, size_t ws_size,
                              hipStream_t stream) {
  const float* x = (const float*)d_in[0];   // [256, 196, 768] f32
  const float* w = (const float*)d_in[1];   // [768, 196] f32
  float* y = (float*)d_out;                 // [256, 196, 768] f32

  // workspace layout: D (bf16 208x224) | Dinv (bf16 208x224) | Wp (f32 768x208)
  __bf16* D  = (__bf16*)d_ws;
  __bf16* Di = D + MR * NP;
  float*  Wp = (float*)((char*)d_ws + (size_t)2 * MR * NP * sizeof(__bf16));

  gen_dct_mats<<<(MR * NP + 255) / 256, 256, 0, stream>>>(D, Di);
  gen_wpad<<<(CDIM * MR + 255) / 256, 256, 0, stream>>>(w, Wp);

  dim3 grid(CDIM / CHUNK, BDIM / GB);       // (12, 64)
  dct_filter_main<<<grid, THREADS, 0, stream>>>(x, Wp, D, Di, y);
}
